// VanillaRNN_20040317403331
// MI455X (gfx1250) — matrix operand order repaired and checked by reading, not compiled, not run on hardware
//
#include <hip/hip_runtime.h>

// ---------------------------------------------------------------------------
// Persistent fused vanilla-RNN kernel for gfx1250 (MI455X).
//   V=E=128, H=256, B=256, S=1024
//   grid = 16 workgroups (one 16-row batch tile each), 256 threads = 8 wave32
//   All weights live in LDS as bf16 (256 KB) -> uses CDNA5 320KB LDS/WGP.
//   Matrix math: v_wmma_f32_16x16x32_bf16 (fp32 accumulate), tanh in fp32.
// ---------------------------------------------------------------------------

typedef __bf16 v16bf __attribute__((ext_vector_type(16)));
typedef __bf16 v8bf  __attribute__((ext_vector_type(8)));
typedef float  v8f   __attribute__((ext_vector_type(8)));

#define Vv 128
#define Ee 128
#define Hh 256
#define Bb 256
#define Ss 1024

// LDS element counts (bf16)
#define L_WIH   (Hh * Ee)         // 32768
#define L_WHH   (Hh * Hh)         // 65536
#define L_WHO   (Vv * Hh)         // 32768
#define L_H     (16 * Hh)         // 4096
#define L_X     (2 * 16 * Ee)     // 4096 (double buffered)
#define SMEM_ELTS (L_WIH + L_WHH + L_WHO + L_H + L_X)     // 139264
#define SMEM_BYTES (SMEM_ELTS * 2)                        // 278528 < 320KB

__device__ __forceinline__ __bf16 f2bf(float f) {
    union { float f; unsigned u; } v; v.f = f;
    unsigned r = v.u + 0x7FFFu + ((v.u >> 16) & 1u);   // round-to-nearest-even
    unsigned short s = (unsigned short)(r >> 16);
    return __builtin_bit_cast(__bf16, s);
}

// A-matrix 16x32 bf16 fragment (ISA 7.12.2): lane holds row M=lane%16;
// half-lanes split K: kbase = (lane>=16)?8:0; VGPR0-3 = K[k0+kbase .. +7],
// VGPR4-7 = K[k0+16+kbase .. +7]. Two 16B LDS loads.
__device__ __forceinline__ v16bf load_a(const __bf16* base, int ldk,
                                        int lrow, int lhi, int k0) {
    const __bf16* p0 = base + lrow * ldk + k0 + lhi * 8;
    v8bf lo = *reinterpret_cast<const v8bf*>(p0);
    v8bf hi = *reinterpret_cast<const v8bf*>(p0 + 16);
    v16bf a;
#pragma unroll
    for (int i = 0; i < 8; ++i) { a[i] = lo[i]; a[8 + i] = hi[i]; }
    return a;
}

// B-matrix 32x16 bf16 fragment: lane holds column N=lane%16; same K split as A:
// kbase = (lane>=16)?8:0; elems 0-7 = K[k0+kbase..+7], elems 8-15 = K[k0+16+kbase..+7].
// Weights stored row-major [n][k] in LDS, so this is 2 16B loads.
__device__ __forceinline__ v16bf load_b(const __bf16* wbase, int ldk,
                                        int nglob, int lhi, int k0) {
    const __bf16* p0 = wbase + nglob * ldk + k0 + lhi * 8;
    v8bf lo = *reinterpret_cast<const v8bf*>(p0);
    v8bf hi = *reinterpret_cast<const v8bf*>(p0 + 16);
    v16bf b;
#pragma unroll
    for (int i = 0; i < 8; ++i) { b[i] = lo[i]; b[8 + i] = hi[i]; }
    return b;
}

extern __shared__ __bf16 smem[];

__global__ __launch_bounds__(256, 1)
void rnn_persistent(const int*   __restrict__ input,    // [B,S]
                    const float* __restrict__ hidden,   // [B,H]
                    const float* __restrict__ emb,      // [V,E]
                    const float* __restrict__ W_ih,     // [H,E]
                    const float* __restrict__ b_ih,     // [H]
                    const float* __restrict__ W_hh,     // [H,H]
                    const float* __restrict__ b_hh,     // [H]
                    const float* __restrict__ W_ho,     // [V,H]
                    const float* __restrict__ b_ho,     // [V]
                    float*       __restrict__ out,      // [B,S,V]
                    float*       __restrict__ h_final)  // [B,H]
{
    __bf16* w_ih_l = smem;                   // [H][E]  row-major (n,k)
    __bf16* w_hh_l = w_ih_l + L_WIH;         // [H][H]
    __bf16* w_ho_l = w_hh_l + L_WHH;         // [V][H]
    __bf16* h_l    = w_ho_l + L_WHO;         // [16][H] (m,k)
    __bf16* x_l    = h_l    + L_H;           // [2][16][E]

    const int tid   = threadIdx.x;
    const int wave  = tid >> 5;              // 0..7 (wave32)
    const int lane  = tid & 31;
    const int lrow  = lane & 15;
    const int lhi   = lane >> 4;
    const int b0    = blockIdx.x * 16;       // batch tile base

    // ---- one-time staging: weights -> bf16 LDS, h0 -> LDS ----
    for (int i = tid; i < L_WIH; i += 256) w_ih_l[i] = f2bf(W_ih[i]);
    for (int i = tid; i < L_WHH; i += 256) w_hh_l[i] = f2bf(W_hh[i]);
    for (int i = tid; i < L_WHO; i += 256) w_ho_l[i] = f2bf(W_ho[i]);
    for (int i = tid; i < L_H;   i += 256) {
        int r = i >> 8, c = i & (Hh - 1);
        h_l[i] = f2bf(hidden[(b0 + r) * Hh + c]);
    }
    // stage x tile for t=0 (embedding gather, fp32 -> bf16)
    {
        int row = tid >> 4, c0 = (tid & 15) * 8;
        int idx = input[(b0 + row) * Ss + 0];
        const float* er = emb + idx * Ee + c0;
        __bf16* xd = x_l + row * Ee + c0;
#pragma unroll
        for (int j = 0; j < 8; ++j) xd[j] = f2bf(er[j]);
    }

    // loop-invariant per-lane biases
    const int   nh0   = (wave * 2 + 0) * 16 + lrow;
    const int   nh1   = (wave * 2 + 1) * 16 + lrow;
    const float biasH0 = b_ih[nh0] + b_hh[nh0];
    const float biasH1 = b_ih[nh1] + b_hh[nh1];
    const int   no0   = wave * 16 + lrow;
    const float biasO = b_ho[no0];

    float hn[2][8];

    for (int t = 0; t < Ss; ++t) {
        __syncthreads();   // x[t&1] and h_l for step t visible

        // stage x tile for t+1 into the other buffer (overlaps WMMA chain)
        if (t + 1 < Ss) {
            int row = tid >> 4, c0 = (tid & 15) * 8;
            int idx = input[(b0 + row) * Ss + (t + 1)];
            const float* er = emb + idx * Ee + c0;
            __bf16* xd = x_l + ((t + 1) & 1) * (16 * Ee) + row * Ee + c0;
#pragma unroll
            for (int j = 0; j < 8; ++j) xd[j] = f2bf(er[j]);
        }

        const __bf16* xcur = x_l + (t & 1) * (16 * Ee);

        // ---- h_new = tanh(x_t @ W_ih^T + h @ W_hh^T + b_ih + b_hh) ----
        // 2 N-tiles of 16 per wave (H=256, 8 waves)
#pragma unroll
        for (int j = 0; j < 2; ++j) {
            const int n0 = (wave * 2 + j) * 16;
            v8f acc = {};
#pragma unroll
            for (int k0 = 0; k0 < Ee; k0 += 32) {                 // input proj
                v16bf a  = load_a(xcur,   Ee, lrow, lhi, k0);
                v16bf bm = load_b(w_ih_l, Ee, n0 + lrow, lhi, k0);
                acc = __builtin_amdgcn_wmma_f32_16x16x32_bf16(
                          false, a, false, bm, (short)0, acc, false, false);
            }
#pragma unroll
            for (int k0 = 0; k0 < Hh; k0 += 32) {                 // recurrence
                v16bf a  = load_a(h_l,    Hh, lrow, lhi, k0);
                v16bf bm = load_b(w_hh_l, Hh, n0 + lrow, lhi, k0);
                acc = __builtin_amdgcn_wmma_f32_16x16x32_bf16(
                          false, a, false, bm, (short)0, acc, false, false);
            }
            const float bias = j ? biasH1 : biasH0;
#pragma unroll
            for (int r = 0; r < 8; ++r) hn[j][r] = tanhf(acc[r] + bias);
        }

        __syncthreads();   // all waves finished reading old h_l

        // write h_new (bf16) back to LDS; acc element r = (M = r + 8*lhi, N = lrow)
#pragma unroll
        for (int j = 0; j < 2; ++j) {
            const int n0 = (wave * 2 + j) * 16;
#pragma unroll
            for (int r = 0; r < 8; ++r)
                h_l[(lhi * 8 + r) * Hh + n0 + lrow] = f2bf(hn[j][r]);
        }

        __syncthreads();   // new h visible to all waves

        // ---- out_t = h_new @ W_ho^T + b_ho  (V=128, 1 N-tile per wave) ----
        {
            const int n0 = wave * 16;
            v8f acc = {};
#pragma unroll
            for (int k0 = 0; k0 < Hh; k0 += 32) {
                v16bf a  = load_a(h_l,    Hh, lrow, lhi, k0);
                v16bf bm = load_b(w_ho_l, Hh, n0 + lrow, lhi, k0);
                acc = __builtin_amdgcn_wmma_f32_16x16x32_bf16(
                          false, a, false, bm, (short)0, acc, false, false);
            }
#pragma unroll
            for (int r = 0; r < 8; ++r) {
                const int b = b0 + lhi * 8 + r;
                out[((size_t)b * Ss + t) * Vv + n0 + lrow] = acc[r] + biasO;
            }
        }

        // final hidden state (fp32, pre-bf16-rounding)
        if (t == Ss - 1) {
#pragma unroll
            for (int j = 0; j < 2; ++j) {
                const int n0 = (wave * 2 + j) * 16;
#pragma unroll
                for (int r = 0; r < 8; ++r)
                    h_final[(size_t)(b0 + lhi * 8 + r) * Hh + n0 + lrow] = hn[j][r];
            }
        }
    }
}

extern "C" void kernel_launch(void* const* d_in, const int* in_sizes, int n_in,
                              void* d_out, int out_size, void* d_ws, size_t ws_size,
                              hipStream_t stream) {
    const int*   input  = (const int*)  d_in[0];
    const float* hidden = (const float*)d_in[1];
    const float* emb    = (const float*)d_in[2];
    const float* W_ih   = (const float*)d_in[3];
    const float* b_ih   = (const float*)d_in[4];
    const float* W_hh   = (const float*)d_in[5];
    const float* b_hh   = (const float*)d_in[6];
    const float* W_ho   = (const float*)d_in[7];
    const float* b_ho   = (const float*)d_in[8];

    float* out     = (float*)d_out;
    float* h_final = out + (size_t)Bb * Ss * Vv;

    // allow >64KB dynamic LDS (CDNA5: 320KB per workgroup)
    (void)hipFuncSetAttribute((const void*)rnn_persistent,
                              hipFuncAttributeMaxDynamicSharedMemorySize,
                              (int)SMEM_BYTES);

    rnn_persistent<<<Bb / 16, 256, SMEM_BYTES, stream>>>(
        input, hidden, emb, W_ih, b_ih, W_hh, b_hh, W_ho, b_ho, out, h_final);
}
